// SparseAttention_65618510348710
// MI455X (gfx1250) — matrix operand order repaired and checked by reading, not compiled, not run on hardware
//
#include <hip/hip_runtime.h>
#include <stdint.h>

typedef _Float16 v16h __attribute__((ext_vector_type(16)));
typedef _Float16 v8h  __attribute__((ext_vector_type(8)));
typedef float    v8f  __attribute__((ext_vector_type(8)));

#define NHEADS 16
#define HD     64        // head dim
#define SQ     4096      // query sequence
#define DMODEL 1024
#define NKEY   1024      // strided keys per head (4096/4)
#define KSTRIDE 4
#define QROWS  128       // q rows per workgroup
#define JCHUNK 32        // keys per iteration
#define SCALE  0.125f    // 1/sqrt(64)

// LDS row strides, padded for bank-conflict avoidance / 16B alignment
#define KSF_STR 68       // f32 staging (async target)
#define VSF_STR 68
#define KS_STR  72       // f16 key-major
#define VT_STR  40       // f16 d-major (transposed V)
#define PW_STR  36       // f32 probability / output staging

static __device__ inline v16h cat16(v8h a, v8h b) {
  return __builtin_shufflevector(a, b, 0,1,2,3,4,5,6,7,8,9,10,11,12,13,14,15);
}
static __device__ inline v8h ld8h(const _Float16* p) {         // 16B-aligned LDS f16
  return *(const v8h*)p;
}
static __device__ inline v8h cvt8(const float* p) {            // 8 f32 -> 8 f16
  float4 a = *(const float4*)p;
  float4 b = *(const float4*)(p + 4);
  v8h r;
  r[0] = (_Float16)a.x; r[1] = (_Float16)a.y; r[2] = (_Float16)a.z; r[3] = (_Float16)a.w;
  r[4] = (_Float16)b.x; r[5] = (_Float16)b.y; r[6] = (_Float16)b.z; r[7] = (_Float16)b.w;
  return r;
}

// --- gfx1250 async global->LDS copy (ASYNCcnt-tracked), ISA §10 / §15.18 ---
static __device__ inline void async_b128(uint32_t lds_off, const float* src) {
  asm volatile("global_load_async_to_lds_b128 %0, %1, off"
               :: "v"(lds_off), "v"((uint64_t)(uintptr_t)src) : "memory");
}
static __device__ inline void wait_async0() {
  asm volatile("s_wait_asynccnt 0" ::: "memory");
}
// issue one 32row x 64col f32 chunk: 512 b128 ops, 2 per thread
static __device__ inline void issue_chunk(const float* srcBase, uint32_t ldsBase,
                                          int ldsStrideF, int tid) {
#pragma unroll
  for (int i = 0; i < 2; ++i) {
    int idx = tid + i * 256;
    int row = idx >> 4, c4 = (idx & 15) << 2;
    uint32_t dst = ldsBase + (uint32_t)(row * ldsStrideF + c4) * 4u;
    async_b128(dst, srcBase + (size_t)row * (KSTRIDE * DMODEL) + c4);
  }
}

__global__ __launch_bounds__(256) void sparse_attn_wmma(
    const float* __restrict__ q, const float* __restrict__ k,
    const float* __restrict__ v, float* __restrict__ outA,
    float* __restrict__ outW)
{
  __shared__ __align__(16) float    KsF[JCHUNK][KSF_STR];   // 8.5 KB async landing (K)
  __shared__ __align__(16) float    VsF[JCHUNK][VSF_STR];   // 8.5 KB async landing (V)
  __shared__ __align__(16) _Float16 Ks[JCHUNK][KS_STR];     // 4.5 KB key-major f16
  __shared__ __align__(16) _Float16 Vt[HD][VT_STR];         // 5 KB   d-major f16
  __shared__ __align__(16) float    Pw[QROWS][PW_STR];      // 18 KB  prob/output staging

  const int nQB  = SQ / QROWS;              // 32
  const int bh   = blockIdx.x / nQB;
  const int qblk = blockIdx.x % nQB;
  const int b    = bh / NHEADS;
  const int h    = bh % NHEADS;
  const int tid  = threadIdx.x;
  const int wave = tid >> 5;
  const int lane = tid & 31;
  const int hl   = lane & 15;
  const bool hi  = lane >= 16;

  const int qbase = qblk * QROWS;
  const float* qB = q + (size_t)b * SQ * DMODEL + h * HD;
  const float* kB = k + (size_t)b * SQ * DMODEL + h * HD;
  const float* vB = v + (size_t)b * SQ * DMODEL + h * HD;

  const uint32_t ksfBase = (uint32_t)(uintptr_t)&KsF[0][0];
  const uint32_t vsfBase = (uint32_t)(uintptr_t)&VsF[0][0];

  // WMMA A operands for this wave's Q rows, built once straight from global.
  // A in contiguous k order (this code's choice): lanes 0-15 -> M=lane, K in {0..15}; lanes 16-31 -> K in {16..31}
  const int abase = hi ? 16 : 0;
  const float* qrow = qB + (size_t)(qbase + wave * 16 + hl) * DMODEL;
  const v16h aq_lo = cat16(cvt8(qrow + abase),      cvt8(qrow + abase + 8));    // d 0..31
  const v16h aq_hi = cat16(cvt8(qrow + abase + 32), cvt8(qrow + abase + 40));   // d 32..63
  const int boff = hi ? 16 : 0;      // B in contiguous k order like A: lane=N, hi half-wave holds K+16

  const v8f vzero = {0.f,0.f,0.f,0.f,0.f,0.f,0.f,0.f};

  // ================= pass 1: row sums of exp(scores) =================
  float rsum[8];
#pragma unroll
  for (int r = 0; r < 8; ++r) rsum[r] = 0.f;

  issue_chunk(kB, ksfBase, KSF_STR, tid);                 // prefetch chunk 0

  for (int j0 = 0; j0 < NKEY; j0 += JCHUNK) {
    wait_async0();
    __syncthreads();
    // convert K chunk f32 -> f16 (once per workgroup)
#pragma unroll
    for (int i = 0; i < 2; ++i) {
      int idx = tid + i * 256;
      int row = idx >> 4, c4 = (idx & 15) << 2;
      float4 f = *(const float4*)&KsF[row][c4];
      Ks[row][c4 + 0] = (_Float16)f.x;
      Ks[row][c4 + 1] = (_Float16)f.y;
      Ks[row][c4 + 2] = (_Float16)f.z;
      Ks[row][c4 + 3] = (_Float16)f.w;
    }
    __syncthreads();
    if (j0 + JCHUNK < NKEY)                                // DMA next chunk under compute
      issue_chunk(kB + (size_t)(j0 + JCHUNK) * (KSTRIDE * DMODEL), ksfBase, KSF_STR, tid);

    const _Float16* kr0 = &Ks[hl][0];
    const _Float16* kr1 = &Ks[16 + hl][0];
    v16h b00 = cat16(ld8h(kr0 + boff),      ld8h(kr0 + boff + 8));
    v16h b01 = cat16(ld8h(kr0 + boff + 32), ld8h(kr0 + boff + 40));
    v16h b10 = cat16(ld8h(kr1 + boff),      ld8h(kr1 + boff + 8));
    v16h b11 = cat16(ld8h(kr1 + boff + 32), ld8h(kr1 + boff + 40));

    v8f c0 = __builtin_amdgcn_wmma_f32_16x16x32_f16(false, aq_lo, false, b00, (short)0, vzero, false, false);
    c0     = __builtin_amdgcn_wmma_f32_16x16x32_f16(false, aq_hi, false, b01, (short)0, c0,    false, false);
    v8f c1 = __builtin_amdgcn_wmma_f32_16x16x32_f16(false, aq_lo, false, b10, (short)0, vzero, false, false);
    c1     = __builtin_amdgcn_wmma_f32_16x16x32_f16(false, aq_hi, false, b11, (short)0, c1,    false, false);

#pragma unroll
    for (int r = 0; r < 8; ++r)
      rsum[r] += __expf(c0[r] * SCALE) + __expf(c1[r] * SCALE);
    __syncthreads();
  }

  // prefetch K+V chunk 0 for pass 2 while reducing
  issue_chunk(kB, ksfBase, KSF_STR, tid);
  issue_chunk(vB, vsfBase, VSF_STR, tid);

  float rinv[8];
#pragma unroll
  for (int r = 0; r < 8; ++r) {
    float s = rsum[r];
    s += __shfl_xor(s, 1, 32);
    s += __shfl_xor(s, 2, 32);
    s += __shfl_xor(s, 4, 32);
    s += __shfl_xor(s, 8, 32);
    rinv[r] = 1.f / s;
  }

  // ================= pass 2: write w (b128), accumulate A = P.V =================
  v8f acc[4];
#pragma unroll
  for (int t = 0; t < 4; ++t) acc[t] = vzero;

  for (int j0 = 0; j0 < NKEY; j0 += JCHUNK) {
    wait_async0();
    __syncthreads();
#pragma unroll
    for (int i = 0; i < 2; ++i) {
      int idx = tid + i * 256;
      int row = idx >> 4, c4 = (idx & 15) << 2;
      float4 f = *(const float4*)&KsF[row][c4];
      Ks[row][c4 + 0] = (_Float16)f.x;
      Ks[row][c4 + 1] = (_Float16)f.y;
      Ks[row][c4 + 2] = (_Float16)f.z;
      Ks[row][c4 + 3] = (_Float16)f.w;
      float4 g = *(const float4*)&VsF[row][c4];
      Vt[c4 + 0][row] = (_Float16)g.x;       // transpose: d-major
      Vt[c4 + 1][row] = (_Float16)g.y;
      Vt[c4 + 2][row] = (_Float16)g.z;
      Vt[c4 + 3][row] = (_Float16)g.w;
    }
    __syncthreads();
    if (j0 + JCHUNK < NKEY) {
      issue_chunk(kB + (size_t)(j0 + JCHUNK) * (KSTRIDE * DMODEL), ksfBase, KSF_STR, tid);
      issue_chunk(vB + (size_t)(j0 + JCHUNK) * (KSTRIDE * DMODEL), vsfBase, VSF_STR, tid);
    }

    const _Float16* kr0 = &Ks[hl][0];
    const _Float16* kr1 = &Ks[16 + hl][0];
    v16h b00 = cat16(ld8h(kr0 + boff),      ld8h(kr0 + boff + 8));
    v16h b01 = cat16(ld8h(kr0 + boff + 32), ld8h(kr0 + boff + 40));
    v16h b10 = cat16(ld8h(kr1 + boff),      ld8h(kr1 + boff + 8));
    v16h b11 = cat16(ld8h(kr1 + boff + 32), ld8h(kr1 + boff + 40));

    v8f c0 = __builtin_amdgcn_wmma_f32_16x16x32_f16(false, aq_lo, false, b00, (short)0, vzero, false, false);
    c0     = __builtin_amdgcn_wmma_f32_16x16x32_f16(false, aq_hi, false, b01, (short)0, c0,    false, false);
    v8f c1 = __builtin_amdgcn_wmma_f32_16x16x32_f16(false, aq_lo, false, b10, (short)0, vzero, false, false);
    c1     = __builtin_amdgcn_wmma_f32_16x16x32_f16(false, aq_hi, false, b11, (short)0, c1,    false, false);

    // normalize and stage probabilities (f32) in LDS
#pragma unroll
    for (int r = 0; r < 8; ++r) {
      int mr = r + (hi ? 8 : 0);
      float p0 = __expf(c0[r] * SCALE) * rinv[r];
      float p1 = __expf(c1[r] * SCALE) * rinv[r];
      Pw[wave * 16 + mr][hl]      = p0;
      Pw[wave * 16 + mr][16 + hl] = p1;
    }

    // A operand for P.V: read this wave's own rows back (same-wave LDS is in-order)
    const float* prow = &Pw[wave * 16 + hl][0];
    v16h ap = cat16(cvt8(prow + abase), cvt8(prow + abase + 8));

#pragma unroll
    for (int t = 0; t < 4; ++t) {
      const _Float16* vrow = &Vt[t * 16 + hl][0];   // B: lane = d column, K = key index
      v16h bv = cat16(ld8h(vrow + boff), ld8h(vrow + boff + 8));
      acc[t] = __builtin_amdgcn_wmma_f32_16x16x32_f16(false, ap, false, bv, (short)0, acc[t], false, false);
    }

    __syncthreads();
    // cooperative wide store of w: 128 rows x 32 cols, b128, fully coalesced
#pragma unroll
    for (int i = 0; i < 4; ++i) {
      int idx = tid + i * 256;
      int row = idx >> 3, c4 = (idx & 7) << 2;
      float4 f = *(const float4*)&Pw[row][c4];
      *(float4*)(outW + ((size_t)bh * SQ + qbase + row) * (size_t)NKEY + j0 + c4) = f;
    }
    __syncthreads();
  }

  // ---------------- write A via Pw, two 32-column halves, b128 stores ----------------
#pragma unroll
  for (int half = 0; half < 2; ++half) {
#pragma unroll
    for (int tt = 0; tt < 2; ++tt) {
#pragma unroll
      for (int r = 0; r < 8; ++r) {
        int mr = r + (hi ? 8 : 0);
        Pw[wave * 16 + mr][tt * 16 + hl] = acc[half * 2 + tt][r];
      }
    }
    __syncthreads();
#pragma unroll
    for (int i = 0; i < 4; ++i) {
      int idx = tid + i * 256;
      int row = idx >> 3, c4 = (idx & 7) << 2;
      float4 f = *(const float4*)&Pw[row][c4];
      *(float4*)(outA + ((size_t)b * SQ + qbase + row) * (size_t)DMODEL + h * HD + half * 32 + c4) = f;
    }
    __syncthreads();
  }
}

extern "C" void kernel_launch(void* const* d_in, const int* in_sizes, int n_in,
                              void* d_out, int out_size, void* d_ws, size_t ws_size,
                              hipStream_t stream) {
  const float* q = (const float*)d_in[0];
  const float* k = (const float*)d_in[1];
  const float* v = (const float*)d_in[2];
  float* outA = (float*)d_out;
  float* outW = outA + (size_t)4 * SQ * DMODEL;   // a is 4*4096*1024 floats
  dim3 grid(4 * NHEADS * (SQ / QROWS));           // 2048 workgroups
  sparse_attn_wmma<<<grid, dim3(256), 0, stream>>>(q, k, v, outA, outW);
}
